// CustomGRULayer_33921651704536
// MI455X (gfx1250) — matrix operand order repaired and checked by reading, not compiled, not run on hardware
//
#include <hip/hip_runtime.h>
#include <hip/hip_bf16.h>
#include <math.h>

// ---------------- problem constants (S,B,I,H fixed by the reference) --------
#define S_LEN   1024
#define BATCH   128
#define IN_DIM  256
#define HID     256
#define G3      768              // 3*H gate columns
#define NWG     16               // workgroups in persistent scan grid
#define NTHREADS 256             // 8 wave32 per workgroup
#define PACK_TILE (8 * 32 * 16)  // bf16 elems per (n-tile): 8 k-tiles * 32 lanes * 16 halves

typedef __attribute__((ext_vector_type(16))) __bf16 v16bf;
typedef __attribute__((ext_vector_type(8)))  __bf16 v8bf;
typedef __attribute__((ext_vector_type(8)))  float  v8f;
typedef __attribute__((ext_vector_type(4)))  float  v4f;

__device__ __forceinline__ v16bf cat8(v8bf lo, v8bf hi) {
  return __builtin_shufflevector(lo, hi, 0,1,2,3,4,5,6,7,8,9,10,11,12,13,14,15);
}

// ---------------------------------------------------------------------------
// Pack weights into WMMA B-operand layout (bf16), convert h0 to bf16, reset
// the grid-barrier counter.  Packed element:
//   dst[((nt*8 + kt)*32 + lane)*16 + h]  where lane = khalf*16 + (n%16),
//   n = nt*16 + lane%16,  k = kt*32 + (lane/16)*8 + h + (h>=8 ? 8 : 0)
// For wih  (x-GEMM):  B[k][n] = wih[n*IN_DIM + k]   (K = input dim)
// For whh_t(h-GEMM):  B[k][n] = whh_t[k*G3 + n]     (K = hidden dim)
// ---------------------------------------------------------------------------
__global__ void gru_pack_kernel(const float* __restrict__ wih,
                                const float* __restrict__ whh_t,
                                const float* __restrict__ h0,
                                __bf16* __restrict__ wih_pack,
                                __bf16* __restrict__ whh_pack,
                                __bf16* __restrict__ h_bf0,
                                unsigned* __restrict__ counter) {
  int idx = blockIdx.x * blockDim.x + threadIdx.x;
  const int PACK_ELEMS = 48 * PACK_TILE;            // 196608 per weight
  if (idx < PACK_ELEMS) {
    int h    = idx & 15;
    int lane = (idx >> 4) & 31;
    int kt   = (idx >> 9) & 7;
    int nt   = idx >> 12;
    int n = nt * 16 + (lane & 15);
    int k = kt * 32 + (lane >> 4) * 8 + h + (h >= 8 ? 8 : 0);
    wih_pack[idx] = (__bf16)wih[n * IN_DIM + k];
    whh_pack[idx] = (__bf16)whh_t[k * G3 + n];
  }
  if (idx < BATCH * HID) h_bf0[idx] = (__bf16)h0[idx];
  if (idx == 0) *counter = 0u;
}

// ---------------------------------------------------------------------------
// One GRU timestep for this lane's fixed 16x16 tiles.  hread/hwrite are the
// raw global kernel arguments (never selected through memory) so the compiler
// keeps GLOBAL addressing (no FLAT, no DScnt coupling).
// ---------------------------------------------------------------------------
struct LaneCtx {
  const __bf16* bxp[3];   // packed Wih tiles for gates r,z,n
  const __bf16* bhp[3];   // packed Whh tiles for gates r,z,n
  const float*  xrow_base;// x + arow*IN_DIM
  float*        h_out;
  unsigned*     counter;
  int           arow;     // A-operand row this lane reads
  int           lhi;      // lane>>4
  int           j;        // hidden unit (output column)
  int           b0;       // first of this lane's 8 batch rows
  float bihr, bhhr, bihz, bhhz, bihn, bhhn;
};

__device__ __forceinline__ void gru_step(const LaneCtx& c, int s,
                                         const __bf16* __restrict__ hread,
                                         __bf16* __restrict__ hwrite,
                                         float (&hprev)[8]) {
  const float*  xrow = c.xrow_base + (size_t)s * (BATCH * IN_DIM);
  const __bf16* hrow = hread + (size_t)c.arow * HID;

  v8f accX[3]; v8f accH[3];
  #pragma unroll
  for (int g = 0; g < 3; ++g) { accX[g] = (v8f)0.0f; accH[g] = (v8f)0.0f; }

  #pragma unroll
  for (int kt = 0; kt < 8; ++kt) {
    const int c0 = kt * 32 + c.lhi * 8;
    // A-operand from x[s] : 16 fp32 -> bf16 (two 8-element K groups)
    v4f f0 = *(const v4f*)(xrow + c0);
    v4f f1 = *(const v4f*)(xrow + c0 + 4);
    v4f f2 = *(const v4f*)(xrow + c0 + 16);
    v4f f3 = *(const v4f*)(xrow + c0 + 20);
    v16bf ax;
    #pragma unroll
    for (int t = 0; t < 4; ++t) {
      ax[t]      = (__bf16)f0[t];  ax[4 + t]  = (__bf16)f1[t];
      ax[8 + t]  = (__bf16)f2[t];  ax[12 + t] = (__bf16)f3[t];
    }
    // A-operand from h (already bf16): two contiguous 16B loads
    v16bf ah = cat8(*(const v8bf*)(hrow + c0), *(const v8bf*)(hrow + c0 + 16));

    #pragma unroll
    for (int g = 0; g < 3; ++g) {
      const __bf16* pb = c.bxp[g] + (size_t)(kt * 32 + (threadIdx.x & 31)) * 16;
      v16bf bw = cat8(*(const v8bf*)pb, *(const v8bf*)(pb + 8));
      accX[g] = __builtin_amdgcn_wmma_f32_16x16x32_bf16(
          false, ax, false, bw, (short)0, accX[g], false, false);
      const __bf16* ph = c.bhp[g] + (size_t)(kt * 32 + (threadIdx.x & 31)) * 16;
      v16bf hw = cat8(*(const v8bf*)ph, *(const v8bf*)(ph + 8));
      accH[g] = __builtin_amdgcn_wmma_f32_16x16x32_bf16(
          false, ah, false, hw, (short)0, accH[g], false, false);
    }
  }

  if (s + 1 < S_LEN)  // warm L2/WGP$ for the next step's x rows
    __builtin_prefetch(xrow + (size_t)BATCH * IN_DIM, 0, 1);

  // ---- GRU gate math (fp32, in registers) + stores ----
  float* outrow = c.h_out + (size_t)s * (BATCH * HID);
  #pragma unroll
  for (int i = 0; i < 8; ++i) {
    float gr = (accX[0][i] + c.bihr) + (accH[0][i] + c.bhhr);
    float gz = (accX[1][i] + c.bihz) + (accH[1][i] + c.bhhz);
    float r  = 1.0f / (1.0f + __expf(-gr));
    float z  = 1.0f / (1.0f + __expf(-gz));
    float n  = tanhf((accX[2][i] + c.bihn) + r * (accH[2][i] + c.bhhn));
    float h  = (1.0f - z) * n + z * hprev[i];
    hprev[i] = h;
    const size_t off = (size_t)(c.b0 + i) * HID + c.j;
    outrow[off] = h;
    hwrite[off] = (__bf16)h;
  }

  // ---- device-wide barrier (monotonic counter, graph-replay safe) ----
  __builtin_amdgcn_fence(__ATOMIC_RELEASE, "agent");
  __syncthreads();
  if (threadIdx.x == 0) {
    __hip_atomic_fetch_add(c.counter, 1u, __ATOMIC_RELAXED, __HIP_MEMORY_SCOPE_AGENT);
    const unsigned target = (unsigned)(NWG * (s + 1));
    while (__hip_atomic_load(c.counter, __ATOMIC_RELAXED,
                             __HIP_MEMORY_SCOPE_AGENT) < target)
      __builtin_amdgcn_s_sleep(1);
  }
  __syncthreads();
  __builtin_amdgcn_fence(__ATOMIC_ACQUIRE, "agent");
}

// ---------------------------------------------------------------------------
// Persistent fused GRU scan.  16 WGs x 256 threads (8 wave32).
//   blockIdx.x = wg   -> owns hidden units [wg*16, wg*16+16)
//   wave (tid>>5)     -> batch tile [wave*16, wave*16+16)
// Time loop unrolled x2 with explicit ping->pong / pong->ping hidden-state
// buffers (pointers stay kernel args => GLOBAL addressing).
// ---------------------------------------------------------------------------
__global__ void __launch_bounds__(NTHREADS)
gru_scan_kernel(const float* __restrict__ x,        // [S,B,I]
                const float* __restrict__ h0,       // [B,H]
                const float* __restrict__ bias_ih,  // [3H]
                const float* __restrict__ bias_hh,  // [3H]
                const __bf16* __restrict__ wih_pack,
                const __bf16* __restrict__ whh_pack,
                __bf16* __restrict__ h_bf0,         // ping (holds h_s for even s)
                __bf16* __restrict__ h_bf1,         // pong
                unsigned* __restrict__ counter,
                float* __restrict__ h_out,          // [S,B,H]
                float* __restrict__ h_final)        // [B,H]
{
  const int wg   = blockIdx.x;          // 0..15 : hidden-unit tile
  const int wave = threadIdx.x >> 5;    // 0..7  : batch tile
  const int lane = threadIdx.x & 31;
  const int lrow = lane & 15;
  const int lhi  = lane >> 4;

  LaneCtx c;
  c.lhi  = lhi;
  c.j    = wg * 16 + lrow;              // hidden unit (C/D layout: n = lane%16)
  c.b0   = wave * 16 + lhi * 8;         // first of 8 batch rows (m = 8*(lane/16)+r)
  c.arow = wave * 16 + lrow;            // A-operand row (A layout: m = lane%16)
  c.xrow_base = x + (size_t)c.arow * IN_DIM;
  c.h_out   = h_out;
  c.counter = counter;

  c.bihr = bias_ih[c.j];            c.bhhr = bias_hh[c.j];
  c.bihz = bias_ih[HID + c.j];      c.bhhz = bias_hh[HID + c.j];
  c.bihn = bias_ih[2 * HID + c.j];  c.bhhn = bias_hh[2 * HID + c.j];

  #pragma unroll
  for (int g = 0; g < 3; ++g) {
    c.bxp[g] = wih_pack + (size_t)(wg + g * NWG) * PACK_TILE;
    c.bhp[g] = whh_pack + (size_t)(wg + g * NWG) * PACK_TILE;
  }

  float hprev[8];
  #pragma unroll
  for (int i = 0; i < 8; ++i) hprev[i] = h0[(size_t)(c.b0 + i) * HID + c.j];

  for (int s = 0; s < S_LEN; s += 2) {
    gru_step(c, s,     h_bf0, h_bf1, hprev);   // reads ping, writes pong
    gru_step(c, s + 1, h_bf1, h_bf0, hprev);   // reads pong, writes ping
  }

  #pragma unroll
  for (int i = 0; i < 8; ++i)
    h_final[(size_t)(c.b0 + i) * HID + c.j] = hprev[i];
}

// ---------------------------------------------------------------------------
extern "C" void kernel_launch(void* const* d_in, const int* in_sizes, int n_in,
                              void* d_out, int out_size, void* d_ws, size_t ws_size,
                              hipStream_t stream) {
  const float* x    = (const float*)d_in[0];   // [S,B,I]
  const float* h0   = (const float*)d_in[1];   // [B,H]
  const float* wih  = (const float*)d_in[2];   // [3H,I]
  const float* whht = (const float*)d_in[3];   // [H,3H]
  const float* bih  = (const float*)d_in[4];   // [3H]
  const float* bhh  = (const float*)d_in[5];   // [3H]

  char* ws = (char*)d_ws;
  __bf16*   wih_pack = (__bf16*)(ws);                     // 393216 B
  __bf16*   whh_pack = (__bf16*)(ws + 393216);            // 393216 B
  __bf16*   h_bf0    = (__bf16*)(ws + 786432);            //  65536 B
  __bf16*   h_bf1    = (__bf16*)(ws + 851968);            //  65536 B
  unsigned* counter  = (unsigned*)(ws + 917504);          //      4 B

  float* h_out   = (float*)d_out;                         // [S,B,H]
  float* h_final = h_out + (size_t)S_LEN * BATCH * HID;   // [B,H]

  gru_pack_kernel<<<768, 256, 0, stream>>>(wih, whht, h0, wih_pack, whh_pack,
                                           h_bf0, counter);
  gru_scan_kernel<<<NWG, NTHREADS, 0, stream>>>(x, h0, bih, bhh,
                                                wih_pack, whh_pack,
                                                h_bf0, h_bf1, counter,
                                                h_out, h_final);
}
